// Extra_Modal_Modeling_62852551410315
// MI455X (gfx1250) — matrix operand order repaired and checked by reading, not compiled, not run on hardware
//
#include <hip/hip_runtime.h>
#include <math.h>

// ---------------------------------------------------------------------------
// Types for CDNA5 WMMA
// ---------------------------------------------------------------------------
typedef __attribute__((ext_vector_type(16))) __bf16 v16bf;
typedef __attribute__((ext_vector_type(8)))  float  v8f;

__device__ __forceinline__ unsigned short f2bf(float f) {
    unsigned int u = __float_as_uint(f);
    unsigned int r = u + 0x7FFFu + ((u >> 16) & 1u);   // round-to-nearest-even
    return (unsigned short)(r >> 16);
}

// ---------------------------------------------------------------------------
// WMMA GEMM:  C[M,N] = A[M,K](bf16) * WT[N,K](bf16)^T + bias[N]   (f32 out)
// One 64x16 output tile per wave (wave32): 4 M-subtiles share one B fragment,
// so each K-step issues 4 v_wmma_f32_16x16x32_bf16 for 10 b128 loads.
// Requires: M % 64 == 0, N % 16 == 0, K % 32 == 0 (true for all call sites).
// A fragment per ISA 16-bit A layout: lane m=l%16, hs=l/16 -> K {hs*8..+7, 16+hs*8..+7}
// B fragment: lane n=l%16, hs=l/16 -> K {hs*8..+7, 16+hs*8..+7} (same as A)
// ---------------------------------------------------------------------------
__global__ void __launch_bounds__(256) gemm_bf16_wmma(
    const unsigned short* __restrict__ A,
    const unsigned short* __restrict__ WT,
    const float* __restrict__ bias,
    float* __restrict__ C,
    int M, int N, int K)
{
    int wave = threadIdx.x >> 5;
    int lane = threadIdx.x & 31;
    int tilesN  = N >> 4;
    int tilesM  = M >> 6;                          // 64-row tiles
    long tile = (long)blockIdx.x * 8 + wave;
    if (tile >= (long)tilesM * tilesN) return;
    int tm = (int)(tile / tilesN);
    int tn = (int)(tile % tilesN);
    int r  = lane & 15;
    int hs = lane >> 4;

    const unsigned short* arow0 = A  + ((long)(tm * 64 +  0 + r)) * K;
    const unsigned short* arow1 = A  + ((long)(tm * 64 + 16 + r)) * K;
    const unsigned short* arow2 = A  + ((long)(tm * 64 + 32 + r)) * K;
    const unsigned short* arow3 = A  + ((long)(tm * 64 + 48 + r)) * K;
    const unsigned short* brow  = WT + ((long)(tn * 16 + r)) * K;

    v8f acc0 = {}, acc1 = {}, acc2 = {}, acc3 = {};
    union Frag { v16bf v; uint4 q[2]; };

    for (int k0 = 0; k0 < K; k0 += 32) {
        Frag b;
        b.q[0] = *(const uint4*)(brow + k0 + hs * 8);
        b.q[1] = *(const uint4*)(brow + k0 + 16 + hs * 8);
        Frag a0, a1, a2, a3;
        a0.q[0] = *(const uint4*)(arow0 + k0 + hs * 8);
        a0.q[1] = *(const uint4*)(arow0 + k0 + 16 + hs * 8);
        a1.q[0] = *(const uint4*)(arow1 + k0 + hs * 8);
        a1.q[1] = *(const uint4*)(arow1 + k0 + 16 + hs * 8);
        a2.q[0] = *(const uint4*)(arow2 + k0 + hs * 8);
        a2.q[1] = *(const uint4*)(arow2 + k0 + 16 + hs * 8);
        a3.q[0] = *(const uint4*)(arow3 + k0 + hs * 8);
        a3.q[1] = *(const uint4*)(arow3 + k0 + 16 + hs * 8);
        if (k0 + 32 < K) {                      // gfx1250 global_prefetch, near scope
            __builtin_prefetch(arow0 + k0 + 32, 0, 3);
            __builtin_prefetch(arow1 + k0 + 32, 0, 3);
            __builtin_prefetch(arow2 + k0 + 32, 0, 3);
            __builtin_prefetch(arow3 + k0 + 32, 0, 3);
        }
        acc0 = __builtin_amdgcn_wmma_f32_16x16x32_bf16(false, a0.v, false, b.v, (short)0, acc0, false, false);
        acc1 = __builtin_amdgcn_wmma_f32_16x16x32_bf16(false, a1.v, false, b.v, (short)0, acc1, false, false);
        acc2 = __builtin_amdgcn_wmma_f32_16x16x32_bf16(false, a2.v, false, b.v, (short)0, acc2, false, false);
        acc3 = __builtin_amdgcn_wmma_f32_16x16x32_bf16(false, a3.v, false, b.v, (short)0, acc3, false, false);
    }

    int n = tn * 16 + r;
    float bv = bias ? bias[n] : 0.f;
    long rowBase = (long)tm * 64 + hs * 8;
#pragma unroll
    for (int v = 0; v < 8; ++v) C[(rowBase +  0 + v) * N + n] = acc0[v] + bv;
#pragma unroll
    for (int v = 0; v < 8; ++v) C[(rowBase + 16 + v) * N + n] = acc1[v] + bv;
#pragma unroll
    for (int v = 0; v < 8; ++v) C[(rowBase + 32 + v) * N + n] = acc2[v] + bv;
#pragma unroll
    for (int v = 0; v < 8; ++v) C[(rowBase + 48 + v) * N + n] = acc3[v] + bv;
}

// ---------------------------------------------------------------------------
// Fused LayerNorm (+optional exact-erf GELU) -> bf16.  One wave per row.
// W in {64,128,256}
// ---------------------------------------------------------------------------
__global__ void __launch_bounds__(256) ln_act_bf16(
    const float* __restrict__ X, const float* __restrict__ g,
    const float* __restrict__ be, unsigned short* __restrict__ Yb,
    long rows, int W, int gelu)
{
    long row = (long)blockIdx.x * 8 + (threadIdx.x >> 5);
    if (row >= rows) return;
    int lane = threadIdx.x & 31;
    const float* x = X + row * (long)W;
    int per = W >> 5;                 // elements per lane (<=8)
    float vals[8];
    float s = 0.f;
    for (int i = 0; i < per; ++i) { vals[i] = x[lane + i * 32]; s += vals[i]; }
    for (int m = 16; m; m >>= 1) s += __shfl_xor(s, m, 32);
    float mean = s / W;
    float vs = 0.f;
    for (int i = 0; i < per; ++i) { float d = vals[i] - mean; vs += d * d; }
    for (int m = 16; m; m >>= 1) vs += __shfl_xor(vs, m, 32);
    float rstd = rsqrtf(vs / W + 1e-5f);
    for (int i = 0; i < per; ++i) {
        int c = lane + i * 32;
        float y = (vals[i] - mean) * rstd * g[c] + be[c];
        if (gelu) y = 0.5f * y * (1.f + erff(y * 0.7071067811865475f));
        Yb[row * (long)W + c] = f2bf(y);
    }
}

// ---------------------------------------------------------------------------
// Elementwise converts
// ---------------------------------------------------------------------------
__global__ void cvt_bf16_k(const float* __restrict__ X, unsigned short* __restrict__ Y, long n)
{
    long i = (long)blockIdx.x * blockDim.x + threadIdx.x;
    if (i < n) Y[i] = f2bf(X[i]);
}
__global__ void cvt_tr_bf16_k(const float* __restrict__ X, unsigned short* __restrict__ Y, int R, int C)
{
    long i = (long)blockIdx.x * blockDim.x + threadIdx.x;
    if (i < (long)R * C) {
        int r = (int)(i / C), c = (int)(i % C);
        Y[(long)c * R + r] = f2bf(X[i]);
    }
}
__global__ void copy_f32_k(const float* __restrict__ X, float* __restrict__ Y, long n)
{
    long i = (long)blockIdx.x * blockDim.x + threadIdx.x;
    if (i < n) Y[i] = X[i];
}

// ---------------------------------------------------------------------------
// Build kv = concat(global, f) with zero pad rows  [B, Lp, D]
// ---------------------------------------------------------------------------
__global__ void build_kv_k(const float* __restrict__ g, const float* __restrict__ f,
                           float* __restrict__ kv, int Bb, int Nn, int Lp, int Dd)
{
    long i = (long)blockIdx.x * blockDim.x + threadIdx.x;
    long tot = (long)Bb * Lp * Dd;
    if (i >= tot) return;
    int d = (int)(i % Dd); long r = i / Dd;
    int j = (int)(r % Lp); int b = (int)(r / Lp);
    float v;
    if (j == 0)            v = g[(long)b * Dd + d];
    else if (j <= Nn)      v = f[((long)b * Nn + (j - 1)) * Dd + d];
    else                   v = 0.f;
    kv[i] = v;
}

// ---------------------------------------------------------------------------
// LayerNorm of a single row (f32 out), <<<1,W>>>, W=128
// ---------------------------------------------------------------------------
__global__ void ln_row(const float* __restrict__ x, const float* __restrict__ g,
                       const float* __restrict__ be, float* __restrict__ y, int W)
{
    __shared__ float red[128];
    int t = threadIdx.x;
    float v = x[t];
    red[t] = v; __syncthreads();
    for (int st = 64; st; st >>= 1) { if (t < st) red[t] += red[t + st]; __syncthreads(); }
    float mean = red[0] / W; __syncthreads();
    float d = v - mean; red[t] = d * d; __syncthreads();
    for (int st = 64; st; st >>= 1) { if (t < st) red[t] += red[t + st]; __syncthreads(); }
    float rstd = rsqrtf(red[0] / W + 1e-5f);
    y[t] = (v - mean) * rstd * g[t] + be[t];
}

// ---------------------------------------------------------------------------
// Query projection: qh[b, o] = q[b]·Wq[o,:] + bq[o]   (Wq = first D rows of Wqkv)
// ---------------------------------------------------------------------------
__global__ void qproj_k(const float* __restrict__ q, int qstride,
                        const float* __restrict__ Wqkv, const float* __restrict__ bqkv,
                        float* __restrict__ qh, int Dd)
{
    int b = blockIdx.x, o = threadIdx.x;
    const float* qr = q + (long)b * qstride;
    const float* w  = Wqkv + (long)o * Dd;
    float s = bqkv[o];
    for (int i = 0; i < Dd; ++i) s += qr[i] * w[i];
    qh[(long)b * Dd + o] = s;
}

// ---------------------------------------------------------------------------
// Output projection: y[b,j] (+)= o[b]·Wo[j,:] + bo[j]
// ---------------------------------------------------------------------------
__global__ void outproj_k(const float* __restrict__ o, const float* __restrict__ Wo,
                          const float* __restrict__ bo, float* __restrict__ y,
                          int Dd, int addTo)
{
    int b = blockIdx.x, j = threadIdx.x;
    const float* orow = o + (long)b * Dd;
    const float* w = Wo + (long)j * Dd;
    float s = bo[j];
    for (int i = 0; i < Dd; ++i) s += orow[i] * w[i];
    if (addTo) y[(long)b * Dd + j] += s; else y[(long)b * Dd + j] = s;
}

// ---------------------------------------------------------------------------
// Single-query multi-head attention per (b,h).  grid = B*nh, block = 256
// kh/vh: [B, Lp, D]; scratch: [B*nh*Lp] f32; qbstride 0 => shared query
// ---------------------------------------------------------------------------
__global__ void __launch_bounds__(256) attn1q_k(
    const float* __restrict__ qh, int qbstride,
    const float* __restrict__ kh, const float* __restrict__ vh,
    float* __restrict__ o, int Lreal, int Lp, int Dd, int nh,
    float scale, float* __restrict__ scratch)
{
    int b = blockIdx.x / nh, h = blockIdx.x % nh;
    int hd = Dd / nh;                               // 32
    const float* K_ = kh + ((long)b * Lp) * Dd + h * hd;
    const float* V_ = vh + ((long)b * Lp) * Dd + h * hd;
    float* sc = scratch + ((long)(b * nh + h)) * Lp;
    __shared__ float red[256];
    __shared__ float qs[32];
    if (threadIdx.x < hd) qs[threadIdx.x] = qh[(long)b * qbstride + h * hd + threadIdx.x];
    __syncthreads();

    float lmax = -1e30f;
    for (int j = threadIdx.x; j < Lreal; j += blockDim.x) {
        const float* kr = K_ + (long)j * Dd;
        float s = 0.f;
        for (int d = 0; d < hd; ++d) s += qs[d] * kr[d];
        s *= scale;
        sc[j] = s;
        lmax = fmaxf(lmax, s);
    }
    red[threadIdx.x] = lmax; __syncthreads();
    for (int st = 128; st; st >>= 1) { if (threadIdx.x < st) red[threadIdx.x] = fmaxf(red[threadIdx.x], red[threadIdx.x + st]); __syncthreads(); }
    float mx = red[0]; __syncthreads();

    float lsum = 0.f;
    for (int j = threadIdx.x; j < Lreal; j += blockDim.x) {
        float e = __expf(sc[j] - mx); sc[j] = e; lsum += e;
    }
    red[threadIdx.x] = lsum; __syncthreads();
    for (int st = 128; st; st >>= 1) { if (threadIdx.x < st) red[threadIdx.x] += red[threadIdx.x + st]; __syncthreads(); }
    float inv = 1.f / red[0]; __syncthreads();

    int d = threadIdx.x % hd, sl = threadIdx.x / hd, nsl = blockDim.x / hd;
    float acc = 0.f;
    for (int j = sl; j < Lreal; j += nsl) acc += sc[j] * V_[(long)j * Dd + d];
    red[threadIdx.x] = acc; __syncthreads();
    if (sl == 0) {
        float t = acc;
        for (int s2 = 1; s2 < nsl; ++s2) t += red[d + s2 * hd];
        o[(long)b * Dd + h * hd + d] = t * inv;
    }
}

// ---------------------------------------------------------------------------
// Full (Linear->LN->GELU)x2->Linear MLP for a handful of rows, f32, 1 block/row
// dims all <= 256; blockDim = 256
// ---------------------------------------------------------------------------
__global__ void __launch_bounds__(256) small_mlp_k(
    const float* __restrict__ X,
    const float* __restrict__ W1, const float* __restrict__ b1,
    const float* __restrict__ g1, const float* __restrict__ be1,
    const float* __restrict__ W2, const float* __restrict__ b2,
    const float* __restrict__ g2, const float* __restrict__ be2,
    const float* __restrict__ W3, const float* __restrict__ b3,
    float* __restrict__ Y,
    int d0, int d1, int d2, int d3, int addTo)
{
    int b = blockIdx.x, t = threadIdx.x;
    __shared__ float xs[256], hb[256], red[256];
    if (t < d0) xs[t] = X[(long)b * d0 + t];
    __syncthreads();

    float h = 0.f;
    if (t < d1) { float s = b1[t]; const float* w = W1 + t;
        for (int i = 0; i < d0; ++i) s += xs[i] * w[(long)i * d1]; h = s; }
    red[t] = (t < d1) ? h : 0.f; __syncthreads();
    for (int st = 128; st; st >>= 1) { if (t < st) red[t] += red[t + st]; __syncthreads(); }
    float mean = red[0] / d1; __syncthreads();
    { float d = (t < d1) ? (h - mean) : 0.f; red[t] = d * d; } __syncthreads();
    for (int st = 128; st; st >>= 1) { if (t < st) red[t] += red[t + st]; __syncthreads(); }
    float rstd = rsqrtf(red[0] / d1 + 1e-5f); __syncthreads();
    if (t < d1) { float y = (h - mean) * rstd * g1[t] + be1[t];
        hb[t] = 0.5f * y * (1.f + erff(y * 0.7071067811865475f)); }
    __syncthreads();

    float h2 = 0.f;
    if (t < d2) { float s = b2[t]; const float* w = W2 + t;
        for (int i = 0; i < d1; ++i) s += hb[i] * w[(long)i * d2]; h2 = s; }
    __syncthreads();
    red[t] = (t < d2) ? h2 : 0.f; __syncthreads();
    for (int st = 128; st; st >>= 1) { if (t < st) red[t] += red[t + st]; __syncthreads(); }
    mean = red[0] / d2; __syncthreads();
    { float d = (t < d2) ? (h2 - mean) : 0.f; red[t] = d * d; } __syncthreads();
    for (int st = 128; st; st >>= 1) { if (t < st) red[t] += red[t + st]; __syncthreads(); }
    rstd = rsqrtf(red[0] / d2 + 1e-5f); __syncthreads();
    if (t < d2) { float y = (h2 - mean) * rstd * g2[t] + be2[t];
        xs[t] = 0.5f * y * (1.f + erff(y * 0.7071067811865475f)); }
    __syncthreads();

    if (t < d3) { float s = b3[t]; const float* w = W3 + t;
        for (int i = 0; i < d2; ++i) s += xs[i] * w[(long)i * d3];
        if (addTo) Y[(long)b * d3 + t] += s; else Y[(long)b * d3 + t] = s; }
}

// ---------------------------------------------------------------------------
// Cosine-sim: sim[b,n] = clip((<fc_n, cc_b>/(|fc_n||cc_b|) + 1)/2, 0, 1)
// one wave per token row, width 64
// ---------------------------------------------------------------------------
__global__ void __launch_bounds__(256) sim_k(
    const float* __restrict__ fc, const float* __restrict__ cc,
    float* __restrict__ sim, long rows, int Nn)
{
    long row = (long)blockIdx.x * 8 + (threadIdx.x >> 5);
    if (row >= rows) return;
    int lane = threadIdx.x & 31;
    int b = (int)(row / Nn);
    float a0 = fc[row * 64 + lane],      a1 = fc[row * 64 + 32 + lane];
    float c0 = cc[(long)b * 64 + lane],  c1 = cc[(long)b * 64 + 32 + lane];
    float na = a0 * a0 + a1 * a1, nc = c0 * c0 + c1 * c1, dp = a0 * c0 + a1 * c1;
    for (int m = 16; m; m >>= 1) {
        na += __shfl_xor(na, m, 32);
        nc += __shfl_xor(nc, m, 32);
        dp += __shfl_xor(dp, m, 32);
    }
    if (lane == 0) {
        float s = dp * rsqrtf(na) * rsqrtf(nc);
        s = (s + 1.f) * 0.5f;
        sim[row] = fminf(fmaxf(s, 0.f), 1.f);
    }
}

// ---------------------------------------------------------------------------
// Exact top-k (desc if invert==0, asc if invert==1) via 4-pass 8-bit radix
// threshold select + tie fill.  1 block per batch. Attention over the selected
// set is permutation-invariant, so output order is irrelevant.
// ---------------------------------------------------------------------------
__device__ __forceinline__ unsigned int okey(float x)
{
    unsigned int u = __float_as_uint(x);
    return (u & 0x80000000u) ? ~u : (u | 0x80000000u);
}

__global__ void __launch_bounds__(256) select_topk_k(
    const float* __restrict__ sim, int Nn, int Kk, int invert, int* __restrict__ idx)
{
    int b = blockIdx.x;
    const float* s = sim + (long)b * Nn;
    __shared__ unsigned int hist[256];
    __shared__ unsigned int sPref, sWant;
    __shared__ int cGT, cEQ;
    if (threadIdx.x == 0) { sPref = 0u; sWant = (unsigned)Kk; }
    __syncthreads();
    for (int pass = 0; pass < 4; ++pass) {
        int shift = 24 - pass * 8;
        hist[threadIdx.x] = 0u;
        __syncthreads();
        unsigned int pref = sPref;
        unsigned int hiMask = (pass == 0) ? 0u : (0xFFFFFFFFu << (shift + 8));
        for (int j = threadIdx.x; j < Nn; j += blockDim.x) {
            unsigned int key = okey(s[j]); if (invert) key = ~key;
            if ((key & hiMask) == (pref & hiMask))
                atomicAdd(&hist[(key >> shift) & 255], 1u);
        }
        __syncthreads();
        if (threadIdx.x == 0) {
            unsigned int acc = 0, want = sWant;
            for (int v = 255; v >= 0; --v) {
                if (acc + hist[v] >= want) {
                    sPref = pref | ((unsigned)v << shift);
                    sWant = want - acc;
                    break;
                }
                acc += hist[v];
            }
        }
        __syncthreads();
    }
    unsigned int pref = sPref, want = sWant;
    if (threadIdx.x == 0) { cGT = 0; cEQ = 0; }
    __syncthreads();
    for (int j = threadIdx.x; j < Nn; j += blockDim.x) {
        unsigned int key = okey(s[j]); if (invert) key = ~key;
        if (key > pref) { int p = atomicAdd(&cGT, 1); idx[(long)b * Kk + p] = j; }
    }
    __syncthreads();
    int base = cGT;
    for (int j = threadIdx.x; j < Nn; j += blockDim.x) {
        unsigned int key = okey(s[j]); if (invert) key = ~key;
        if (key == pref) {
            int p = atomicAdd(&cEQ, 1);
            if (p < (int)want) idx[(long)b * Kk + base + p] = j;
        }
    }
}

// ---------------------------------------------------------------------------
// Row gather (bf16): dst[b,j,:] = src[b, idx[b,j], :]
// ---------------------------------------------------------------------------
__global__ void gather_bf16_k(const unsigned short* __restrict__ src, const int* __restrict__ idx,
                              unsigned short* __restrict__ dst, int Bb, int Nn, int Kk, int Dd)
{
    long i = (long)blockIdx.x * blockDim.x + threadIdx.x;
    long tot = (long)Bb * Kk * Dd;
    if (i >= tot) return;
    int d = (int)(i % Dd); long r = i / Dd;
    int j = (int)(r % Kk); int b = (int)(r / Kk);
    dst[i] = src[((long)b * Nn + idx[(long)b * Kk + j]) * Dd + d];
}

// ===========================================================================
// Host orchestration
// ===========================================================================
extern "C" void kernel_launch(void* const* d_in, const int* in_sizes, int n_in,
                              void* d_out, int out_size, void* d_ws, size_t ws_size,
                              hipStream_t stream)
{
    (void)in_sizes; (void)n_in; (void)out_size; (void)ws_size;

    const int   Bb = 8, Nn = 16384, Mm = 2048, Dd = 128, TOKk = 2048, NHh = 4;
    const long  T   = (long)Bb * Nn;        // 131072 tokens
    const long  TC  = (long)Bb * Mm;        // 16384 tokens
    const int   Lp  = 16448;                // (N+1) rounded to 64 (for 64-row GEMM tiles)
    const long  LPB = (long)Bb * Lp;        // padded kv rows
    const float scale = 0.17677669529663687f;   // 1/sqrt(32)

    auto P = [&](int i) { return (const float*)d_in[i]; };

    // ---- input pointer map (setup_inputs dict traversal order) ----
    const float* feats  = P(0);
    const float* gfeat  = P(1);
    const float* crossf = P(2);
    // feat_enc
    const float *W1fe=P(3),*b1fe=P(4),*g1fe=P(5),*be1fe=P(6),
                *W2fe=P(7),*b2fe=P(8),*g2fe=P(9),*be2fe=P(10),
                *W3fe=P(11),*b3fe=P(12);
    // cross_enc
    const float *W1ce=P(13),*b1ce=P(14),*g1ce=P(15),*be1ce=P(16),
                *W2ce=P(17),*b2ce=P(18),*g2ce=P(19),*be2ce=P(20),
                *W3ce=P(21),*b3ce=P(22);
    // consist_feat (128,256,128,64)
    const float *W1cf=P(23),*b1cf=P(24),*g1cf=P(25),*be1cf=P(26),
                *W2cf=P(27),*b2cf=P(28),*g2cf=P(29),*be2cf=P(30),
                *W3cf=P(31),*b3cf=P(32);
    // mha_cross
    const float *Wqkv_c=P(33),*bqkv_c=P(34),*Wo_c=P(35),*bo_c=P(36);
    const float *lncg=P(37),*lncb=P(38),*clsc=P(39);
    // mha_feat
    const float *Wqkv_f=P(40),*bqkv_f=P(41),*Wo_f=P(42),*bo_f=P(43);
    const float *lnfg=P(44),*lnfb=P(45),*clsf=P(46);
    // mha1, mlp1
    const float *Wqkv1=P(47),*bqkv1=P(48),*Wo1=P(49),*bo1=P(50);
    const float *W1m1=P(51),*b1m1=P(52),*g1m1=P(53),*be1m1=P(54),
                *W2m1=P(55),*b2m1=P(56),*g2m1=P(57),*be2m1=P(58),
                *W3m1=P(59),*b3m1=P(60);
    // mha2, mlp2
    const float *Wqkv2=P(61),*bqkv2=P(62),*Wo2=P(63),*bo2=P(64);
    const float *W1m2=P(65),*b1m2=P(66),*g1m2=P(67),*be1m2=P(68),
                *W2m2=P(69),*b2m2=P(70),*g2m2=P(71),*be2m2=P(72),
                *W3m2=P(73),*b3m2=P(74);

    // ---- output layout: feat_aggr [B,1,D] | sim [B,N] | fc [B,N,64] ----
    float* out_fa  = (float*)d_out;
    float* out_sim = out_fa + (long)Bb * Dd;
    float* out_fc  = out_sim + T;

    // ---- workspace arena ----
    char* base = (char*)d_ws;
    size_t off = 0;
    auto alloc = [&](size_t bytes) -> char* {
        char* p = base + off;
        off = (off + bytes + 255) & ~(size_t)255;
        return p;
    };
    unsigned short* fb = (unsigned short*)alloc(T * 128 * 2);     // f as bf16 (persist)
    float*          ff = (float*)alloc(T * 128 * 4);              // f as f32  (persist)
    size_t aA = (size_t)2 * LPB * 128 * 4;                        // kh+vh for mha_feat
    if (aA < (size_t)T * 256 * 4) aA = (size_t)T * 256 * 4;
    char* arenaA = alloc(aA);                                     // h1 f32 / kh+vh
    char* arenaB = alloc((size_t)T * 256 * 4);                    // xb / h2 f32 / kv
    char* arenaC = alloc((size_t)T * 256 * 2);                    // hb bf16 / kvb / patches

    // bf16 weight buffers
    unsigned short* wt1_fe = (unsigned short*)alloc(256*128*2);
    unsigned short* wt2_fe = (unsigned short*)alloc(256*256*2);
    unsigned short* wt3_fe = (unsigned short*)alloc(128*256*2);
    unsigned short* wt1_ce = (unsigned short*)alloc(256*128*2);
    unsigned short* wt2_ce = (unsigned short*)alloc(256*256*2);
    unsigned short* wt3_ce = (unsigned short*)alloc(128*256*2);
    unsigned short* wt1_cf = (unsigned short*)alloc(256*128*2);
    unsigned short* wt2_cf = (unsigned short*)alloc(128*256*2);
    unsigned short* wt3_cf = (unsigned short*)alloc(64*128*2);
    unsigned short* wkC = (unsigned short*)alloc(128*128*2);
    unsigned short* wvC = (unsigned short*)alloc(128*128*2);
    unsigned short* wkF = (unsigned short*)alloc(128*128*2);
    unsigned short* wvF = (unsigned short*)alloc(128*128*2);
    unsigned short* wk1 = (unsigned short*)alloc(128*128*2);
    unsigned short* wv1 = (unsigned short*)alloc(128*128*2);
    unsigned short* wk2 = (unsigned short*)alloc(128*128*2);
    unsigned short* wv2 = (unsigned short*)alloc(128*128*2);

    // small f32 buffers
    float* lncls_c = (float*)alloc(128*4);
    float* qh_c    = (float*)alloc(128*4);
    float* o_c     = (float*)alloc(Bb*128*4);
    float* FAC     = (float*)alloc(Bb*128*4);   // feat_aggr_cross
    float* ccb     = (float*)alloc(Bb*64*4);
    float* lncls_f = (float*)alloc(128*4);
    float* qh_f    = (float*)alloc(128*4);
    float* o_f     = (float*)alloc(Bb*128*4);
    float* FA      = (float*)alloc(Bb*128*4);   // feat_aggr (residual stream)
    float* qh1     = (float*)alloc(Bb*128*4);
    float* o1      = (float*)alloc(Bb*128*4);
    float* mo1     = (float*)alloc(Bb*128*4);
    float* qh2     = (float*)alloc(Bb*128*4);
    float* o2      = (float*)alloc(Bb*128*4);
    float* mo2     = (float*)alloc(Bb*128*4);
    float* scores  = (float*)alloc((size_t)Bb*NHh*Lp*4);
    int*   idx_hi  = (int*)alloc((size_t)Bb*TOKk*4);
    int*   idx_lo  = (int*)alloc((size_t)Bb*TOKk*4);

    // ---- launch helpers ----
    auto gemm = [&](const unsigned short* A, const unsigned short* WT, const float* bias,
                    float* C, long M, int N, int K) {
        long tiles = (M / 64) * (N / 16);
        int blocks = (int)((tiles + 7) / 8);
        gemm_bf16_wmma<<<blocks, 256, 0, stream>>>(A, WT, bias, C, (int)M, N, K);
    };
    auto lnact = [&](const float* X, const float* g, const float* be,
                     unsigned short* Y, long rows, int W2, int gelu) {
        int blocks = (int)((rows + 7) / 8);
        ln_act_bf16<<<blocks, 256, 0, stream>>>(X, g, be, Y, rows, W2, gelu);
    };
    auto cvt = [&](const float* X, unsigned short* Y, long n) {
        cvt_bf16_k<<<(int)((n + 255) / 256), 256, 0, stream>>>(X, Y, n);
    };
    auto cvtT = [&](const float* X, unsigned short* Y, int R, int C) {
        long n = (long)R * C;
        cvt_tr_bf16_k<<<(int)((n + 255) / 256), 256, 0, stream>>>(X, Y, R, C);
    };

    // ---- 0) weight conversion (transpose row-major (K,N) -> WT (N,K) bf16) ----
    cvtT(W1fe, wt1_fe, 128, 256);  cvtT(W2fe, wt2_fe, 256, 256);  cvtT(W3fe, wt3_fe, 256, 128);
    cvtT(W1ce, wt1_ce, 128, 256);  cvtT(W2ce, wt2_ce, 256, 256);  cvtT(W3ce, wt3_ce, 256, 128);
    cvtT(W1cf, wt1_cf, 128, 256);  cvtT(W2cf, wt2_cf, 256, 128);  cvtT(W3cf, wt3_cf, 128, 64);
    cvt(Wqkv_c + 128*128, wkC, 128*128);  cvt(Wqkv_c + 256*128, wvC, 128*128);
    cvt(Wqkv_f + 128*128, wkF, 128*128);  cvt(Wqkv_f + 256*128, wvF, 128*128);
    cvt(Wqkv1  + 128*128, wk1, 128*128);  cvt(Wqkv1  + 256*128, wv1, 128*128);
    cvt(Wqkv2  + 128*128, wk2, 128*128);  cvt(Wqkv2  + 256*128, wv2, 128*128);

    // ---- A) f = feat_enc(feats)   [T,128] ----
    cvt(feats, (unsigned short*)arenaB, T * 128);
    gemm((unsigned short*)arenaB, wt1_fe, b1fe, (float*)arenaA, T, 256, 128);
    lnact((float*)arenaA, g1fe, be1fe, (unsigned short*)arenaC, T, 256, 1);
    gemm((unsigned short*)arenaC, wt2_fe, b2fe, (float*)arenaB, T, 256, 256);
    lnact((float*)arenaB, g2fe, be2fe, (unsigned short*)arenaC, T, 256, 1);
    gemm((unsigned short*)arenaC, wt3_fe, b3fe, ff, T, 128, 256);
    cvt(ff, fb, T * 128);

    // ---- B) c = cross_enc(cross_feat); mha_cross -> FAC ----
    cvt(crossf, (unsigned short*)arenaB, TC * 128);
    gemm((unsigned short*)arenaB, wt1_ce, b1ce, (float*)arenaA, TC, 256, 128);
    lnact((float*)arenaA, g1ce, be1ce, (unsigned short*)arenaC, TC, 256, 1);
    gemm((unsigned short*)arenaC, wt2_ce, b2ce, (float*)arenaB, TC, 256, 256);
    lnact((float*)arenaB, g2ce, be2ce, (unsigned short*)arenaC, TC, 256, 1);
    float* cbuf = (float*)(arenaB + (size_t)TC * 256 * 4);
    gemm((unsigned short*)arenaC, wt3_ce, b3ce, cbuf, TC, 128, 256);
    lnact(cbuf, lncg, lncb, (unsigned short*)arenaC, TC, 128, 0);       // lnc(c) bf16
    float* khc = (float*)arenaA;
    float* vhc = (float*)(arenaA + (size_t)TC * 128 * 4);
    gemm((unsigned short*)arenaC, wkC, bqkv_c + 128, khc, TC, 128, 128);
    gemm((unsigned short*)arenaC, wvC, bqkv_c + 256, vhc, TC, 128, 128);
    ln_row<<<1, 128, 0, stream>>>(clsc, lncg, lncb, lncls_c, 128);
    qproj_k<<<1, 128, 0, stream>>>(lncls_c, 0, Wqkv_c, bqkv_c, qh_c, 128);
    attn1q_k<<<Bb * NHh, 256, 0, stream>>>(qh_c, 0, khc, vhc, o_c, Mm, Mm, Dd, NHh, scale, scores);
    outproj_k<<<Bb, 128, 0, stream>>>(o_c, Wo_c, bo_c, FAC, 128, 0);

    // ---- D) fc = consist(f) -> d_out ; cc = consist(FAC) ; sim ----
    gemm(fb, wt1_cf, b1cf, (float*)arenaA, T, 256, 128);
    lnact((float*)arenaA, g1cf, be1cf, (unsigned short*)arenaC, T, 256, 1);
    gemm((unsigned short*)arenaC, wt2_cf, b2cf, (float*)arenaB, T, 128, 256);
    lnact((float*)arenaB, g2cf, be2cf, (unsigned short*)arenaC, T, 128, 1);
    gemm((unsigned short*)arenaC, wt3_cf, b3cf, out_fc, T, 64, 128);
    small_mlp_k<<<Bb, 256, 0, stream>>>(FAC, W1cf, b1cf, g1cf, be1cf,
                                        W2cf, b2cf, g2cf, be2cf, W3cf, b3cf,
                                        ccb, 128, 256, 128, 64, 0);
    sim_k<<<(int)(T / 8), 256, 0, stream>>>(out_fc, ccb, out_sim, T, Nn);

    // ---- E) mha_feat over kv=concat(global,f) -> FA ----
    {
        long tot = LPB * 128;
        build_kv_k<<<(int)((tot + 255) / 256), 256, 0, stream>>>(gfeat, ff, (float*)arenaB, Bb, Nn, Lp, Dd);
    }
    lnact((float*)arenaB, lnfg, lnfb, (unsigned short*)arenaC, LPB, 128, 0);  // lnf(kv) bf16
    float* khf = (float*)arenaA;
    float* vhf = (float*)(arenaA + (size_t)LPB * 128 * 4);
    gemm((unsigned short*)arenaC, wkF, bqkv_f + 128, khf, LPB, 128, 128);
    gemm((unsigned short*)arenaC, wvF, bqkv_f + 256, vhf, LPB, 128, 128);
    ln_row<<<1, 128, 0, stream>>>(clsf, lnfg, lnfb, lncls_f, 128);
    qproj_k<<<1, 128, 0, stream>>>(lncls_f, 0, Wqkv_f, bqkv_f, qh_f, 128);
    attn1q_k<<<Bb * NHh, 256, 0, stream>>>(qh_f, 0, khf, vhf, o_f, Nn + 1, Lp, Dd, NHh, scale, scores);
    outproj_k<<<Bb, 128, 0, stream>>>(o_f, Wo_f, bo_f, FA, 128, 0);

    // ---- F) top-2048 / bottom-2048 selection + gathers ----
    select_topk_k<<<Bb, 256, 0, stream>>>(out_sim, Nn, TOKk, 0, idx_hi);
    select_topk_k<<<Bb, 256, 0, stream>>>(out_sim, Nn, TOKk, 1, idx_lo);
    unsigned short* simp   = (unsigned short*)arenaC;
    unsigned short* unsimp = (unsigned short*)(arenaC + (size_t)Bb * TOKk * 128 * 2);
    {
        long tot = (long)Bb * TOKk * 128;
        gather_bf16_k<<<(int)((tot + 255) / 256), 256, 0, stream>>>(fb, idx_hi, simp,   Bb, Nn, TOKk, Dd);
        gather_bf16_k<<<(int)((tot + 255) / 256), 256, 0, stream>>>(fb, idx_lo, unsimp, Bb, Nn, TOKk, Dd);
    }

    // ---- G) mha1 + mlp1 residual; mha2 + mlp2 residual ----
    long BT = (long)Bb * TOKk;
    float* kh1 = (float*)arenaA;
    float* vh1 = (float*)(arenaA + (size_t)BT * 128 * 4);
    gemm(simp, wk1, bqkv1 + 128, kh1, BT, 128, 128);
    gemm(simp, wv1, bqkv1 + 256, vh1, BT, 128, 128);
    qproj_k<<<Bb, 128, 0, stream>>>(FA, 128, Wqkv1, bqkv1, qh1, 128);
    attn1q_k<<<Bb * NHh, 256, 0, stream>>>(qh1, 128, kh1, vh1, o1, TOKk, TOKk, Dd, NHh, scale, scores);
    outproj_k<<<Bb, 128, 0, stream>>>(o1, Wo1, bo1, mo1, 128, 0);
    small_mlp_k<<<Bb, 256, 0, stream>>>(mo1, W1m1, b1m1, g1m1, be1m1,
                                        W2m1, b2m1, g2m1, be2m1, W3m1, b3m1,
                                        FA, 128, 256, 256, 128, 1);

    float* kh2 = (float*)arenaA;
    float* vh2 = (float*)(arenaA + (size_t)BT * 128 * 4);
    gemm(unsimp, wk2, bqkv2 + 128, kh2, BT, 128, 128);
    gemm(unsimp, wv2, bqkv2 + 256, vh2, BT, 128, 128);
    qproj_k<<<Bb, 128, 0, stream>>>(FA, 128, Wqkv2, bqkv2, qh2, 128);
    attn1q_k<<<Bb * NHh, 256, 0, stream>>>(qh2, 128, kh2, vh2, o2, TOKk, TOKk, Dd, NHh, scale, scores);
    outproj_k<<<Bb, 128, 0, stream>>>(o2, Wo2, bo2, mo2, 128, 0);
    small_mlp_k<<<Bb, 256, 0, stream>>>(mo2, W1m2, b1m2, g1m2, be1m2,
                                        W2m2, b2m2, g2m2, be2m2, W3m2, b3m2,
                                        FA, 128, 256, 256, 128, 1);

    copy_f32_k<<<4, 256, 0, stream>>>(FA, out_fa, (long)Bb * Dd);
}
